// RGCN_41369124995616
// MI455X (gfx1250) — matrix operand order repaired and checked by reading, not compiled, not run on hardware
//
#include <hip/hip_runtime.h>
#include <hip/hip_bf16.h>
#include <stdint.h>

#define N_NODES  50000
#define N_EDGES  800000
#define N_RELS   65
#define D        64
#define N_GRAPHS 512
#define GROUP    4          // tiles (of 16 edges) per wave; B stays in registers

typedef __attribute__((ext_vector_type(16))) __bf16 v16bf;
typedef __attribute__((ext_vector_type(8)))  float  v8f;

union Frag16 {
    v16bf v;
    uint32_t u[8];
    unsigned short h[16];
};

__device__ __forceinline__ unsigned short f2bf(float f) {
    uint32_t x = __float_as_uint(f);
    uint32_t r = x + 0x7FFFu + ((x >> 16) & 1u);   // round-to-nearest-even
    return (unsigned short)(r >> 16);
}

// ---------------- utility kernels ----------------
__global__ void k_zero_i32(int* p, int n) {
    int i = blockIdx.x * blockDim.x + threadIdx.x;
    if (i < n) p[i] = 0;
}
__global__ void k_zero_f32(float* p, int n) {
    int i = blockIdx.x * blockDim.x + threadIdx.x;
    if (i < n) p[i] = 0.0f;
}
__global__ void k_hist(const int* __restrict__ et, int* __restrict__ counts) {
    int i = blockIdx.x * blockDim.x + threadIdx.x;
    if (i < N_EDGES) atomicAdd(&counts[et[i]], 1);
}
__global__ void k_scan(const int* __restrict__ counts, int* __restrict__ reloff,
                       int* __restrict__ gtileoff, int* __restrict__ total) {
    if (threadIdx.x == 0 && blockIdx.x == 0) {
        int off = 0, goff = 0;
        for (int r = 0; r < N_RELS; ++r) {
            reloff[r] = off;  gtileoff[r] = goff;
            off  += counts[r];
            goff += (counts[r] + (16 * GROUP - 1)) >> 6;   // 64-edge groups
        }
        reloff[N_RELS]   = off;
        gtileoff[N_RELS] = goff;
        *total = goff;
    }
}
__global__ void k_scatter(const int* __restrict__ et, const int* __restrict__ reloff,
                          int* __restrict__ cursor, int* __restrict__ sorted) {
    int e = blockIdx.x * blockDim.x + threadIdx.x;
    if (e < N_EDGES) {
        int r = et[e];
        int pos = reloff[r] + atomicAdd(&cursor[r], 1);
        sorted[pos] = e;
    }
}
__global__ void k_cvt_bf16(const float* __restrict__ x, unsigned short* __restrict__ y, int n) {
    int i = blockIdx.x * blockDim.x + threadIdx.x;
    if (i < n) y[i] = f2bf(x[i]);
}

// Swizzle W (R x 64 x 64, row-major [d][k]) into the exact WMMA B-fragment
// register image: [r][ks(2)][nt(4)][lane(32)][j(16)] bf16, so a fragment load
// is two contiguous b128 per lane (coalesced, no sub-word shuffles).
//   element = W[r][ d = ks*32 + (lane>=16 ? 8:0) + (j&7) + (j>=8 ? 16:0) ][ k = nt*16 + (lane&15) ]
__global__ void k_swizzle_W(const float* __restrict__ Wf, unsigned short* __restrict__ Wswz) {
    int i = blockIdx.x * blockDim.x + threadIdx.x;      // over N_RELS*4096
    if (i >= N_RELS * D * D) return;
    int r    = i >> 12;
    int x    = i & 4095;
    int j    = x & 15;
    int lane = (x >> 4) & 31;
    int nt   = (x >> 9) & 3;
    int ks   = (x >> 11) & 1;
    int dk   = ks * 32 + ((lane & 16) ? 8 : 0) + (j & 7) + ((j & 8) ? 16 : 0);    // K (input dim d)
    int nn   = nt * 16 + (lane & 15);                   // N (output dim k)
    Wswz[i]  = f2bf(Wf[(size_t)r * D * D + dk * D + nn]);
}

__global__ void k_bias_relu_cvt(const float* __restrict__ agg, const float* __restrict__ b,
                                float* __restrict__ h, unsigned short* __restrict__ hbf) {
    int i = blockIdx.x * blockDim.x + threadIdx.x;
    if (i < N_NODES * D) {
        float v = agg[i] + b[i & (D - 1)];
        v = fmaxf(v, 0.0f);
        h[i] = v;
        hbf[i] = f2bf(v);
    }
}
__global__ void k_pool(const float* __restrict__ h, const int* __restrict__ gid,
                       float* __restrict__ g) {
    int i = blockIdx.x * blockDim.x + threadIdx.x;
    if (i < N_NODES * D) {
        int n = i >> 6, d = i & (D - 1);
        atomicAdd(&g[gid[n] * D + d], h[i]);
    }
}

// ---------------- RGCN edge-group WMMA kernel ----------------
// One wave32 per group of GROUP=4 tiles (64 relation-sorted edges).
// All 8 B fragments (64x64 bf16 relation weight) live in registers for the
// whole group. Per tile: gather A (16x64 bf16), 8 x v_wmma_f32_16x16x32_bf16,
// atomic scatter of the 16x64 f32 result into agg[dst].
__launch_bounds__(128)
__global__ void k_rgcn_wmma(const unsigned short* __restrict__ hbf,
                            const unsigned short* __restrict__ Wswz,
                            const int* __restrict__ sorted,
                            const int* __restrict__ srcA,
                            const int* __restrict__ dstA,
                            const int* __restrict__ reloff,
                            const int* __restrict__ gtileoff,
                            const int* __restrict__ total,
                            float* __restrict__ agg)
{
    const int lane = threadIdx.x & 31;
    const int wave = threadIdx.x >> 5;
    const int g = blockIdx.x * 4 + wave;
    if (g >= *total) return;                 // wave-uniform exit

    // locate relation segment for this group (66-entry prefix, wave-uniform)
    int r = 0;
    while (r < N_RELS - 1 && gtileoff[r + 1] <= g) ++r;
    const int lg    = g - gtileoff[r];
    const int gbase = reloff[r] + lg * (16 * GROUP);
    const int end   = reloff[r + 1];

    // load all 8 B fragments (pre-swizzled): 2 coalesced b128 each
    Frag16 bf[2][4];
    {
        const uint4* bw = reinterpret_cast<const uint4*>(Wswz + (size_t)r * D * D);
        #pragma unroll
        for (int ks = 0; ks < 2; ++ks)
            #pragma unroll
            for (int nt = 0; nt < 4; ++nt) {
                const int fi = ((ks * 4 + nt) * 32 + lane) * 2;
                uint4 q0 = bw[fi];
                uint4 q1 = bw[fi + 1];
                bf[ks][nt].u[0] = q0.x; bf[ks][nt].u[1] = q0.y;
                bf[ks][nt].u[2] = q0.z; bf[ks][nt].u[3] = q0.w;
                bf[ks][nt].u[4] = q1.x; bf[ks][nt].u[5] = q1.y;
                bf[ks][nt].u[6] = q1.z; bf[ks][nt].u[7] = q1.w;
            }
    }

    const int m    = lane & 15;              // edge row within tile
    const int koff = (lane < 16) ? 0 : 8;    // A-fragment K sub-offset (ISA layout)
    const int n    = lane & 15;              // C column within 16-wide tile
    const int hi   = lane >> 4;

    for (int tt = 0; tt < GROUP; ++tt) {
        const int base = gbase + tt * 16;
        if (base >= end) break;              // wave-uniform

        const int eidx  = base + m;
        const bool vld  = (eidx < end);
        const int e     = vld ? sorted[eidx] : sorted[base];
        const int snode = srcA[e];
        const int dnode = dstA[e];

        v8f c[4];
        #pragma unroll
        for (int nt = 0; nt < 4; ++nt)
            #pragma unroll
            for (int j = 0; j < 8; ++j) c[nt][j] = 0.0f;

        #pragma unroll
        for (int ks = 0; ks < 2; ++ks) {     // K = ks*32 .. ks*32+31
            Frag16 a;
            if (vld) {
                // lane holds row m: halves K+[0..8) and K+[16..24) (offset koff)
                const uint4* p = reinterpret_cast<const uint4*>(
                    hbf + (size_t)snode * D + ks * 32 + koff);
                uint4 t0 = p[0];
                uint4 t1 = p[2];
                a.u[0] = t0.x; a.u[1] = t0.y; a.u[2] = t0.z; a.u[3] = t0.w;
                a.u[4] = t1.x; a.u[5] = t1.y; a.u[6] = t1.z; a.u[7] = t1.w;
            } else {
                #pragma unroll
                for (int j = 0; j < 8; ++j) a.u[j] = 0u;   // zero rows past end
            }
            #pragma unroll
            for (int nt = 0; nt < 4; ++nt)
                c[nt] = __builtin_amdgcn_wmma_f32_16x16x32_bf16(
                    false, a.v, false, bf[ks][nt].v, (short)0, c[nt], false, false);
        }

        // scatter C: VGPR v holds row M = v + hi*8, column = nt*16 + n
        #pragma unroll
        for (int v = 0; v < 8; ++v) {
            const int M = v + hi * 8;
            if (base + M < end) {
                const int dn = __shfl(dnode, M, 32);
                #pragma unroll
                for (int nt = 0; nt < 4; ++nt)
                    atomicAdd(&agg[(size_t)dn * D + nt * 16 + n], c[nt][v]);
            }
        }
    }
}

// ---------------- tiny MLP head (512x64, negligible FLOPs) ----------------
__global__ void k_mlp(const float* __restrict__ g,
                      const float* __restrict__ W1, const float* __restrict__ b1,
                      const float* __restrict__ W2, const float* __restrict__ b2,
                      const float* __restrict__ W3, const float* __restrict__ b3,
                      const float* __restrict__ pW, const float* __restrict__ pb,
                      float* __restrict__ out)
{
    __shared__ float v[D];
    const int gi  = blockIdx.x;
    const int tid = threadIdx.x;
    v[tid] = g[gi * D + tid];
    __syncthreads();
    const float* Ws[3] = {W1, W2, W3};
    const float* bs[3] = {b1, b2, b3};
    for (int l = 0; l < 3; ++l) {
        float acc = bs[l][tid];
        for (int d = 0; d < D; ++d) acc += v[d] * Ws[l][d * D + tid];
        acc = fmaxf(acc, 0.0f);
        __syncthreads();
        v[tid] = acc;
        __syncthreads();
    }
    if (tid < 2) {
        float acc = pb[tid];
        for (int d = 0; d < D; ++d) acc += v[d] * pW[d * 2 + tid];
        out[gi * 2 + tid] = acc;
    }
}

// ---------------- launch ----------------
extern "C" void kernel_launch(void* const* d_in, const int* in_sizes, int n_in,
                              void* d_out, int out_size, void* d_ws, size_t ws_size,
                              hipStream_t stream)
{
    const float* node_feats = (const float*)d_in[0];
    const int*   etypes     = (const int*)d_in[1];
    const int*   src        = (const int*)d_in[2];
    const int*   dst        = (const int*)d_in[3];
    const int*   gid        = (const int*)d_in[4];
    const float* Wr[3] = {(const float*)d_in[5], (const float*)d_in[7], (const float*)d_in[9]};
    const float* br[3] = {(const float*)d_in[6], (const float*)d_in[8], (const float*)d_in[10]};
    const float* fcW[3] = {(const float*)d_in[11], (const float*)d_in[13], (const float*)d_in[15]};
    const float* fcb[3] = {(const float*)d_in[12], (const float*)d_in[14], (const float*)d_in[16]};
    const float* pW = (const float*)d_in[17];
    const float* pb = (const float*)d_in[18];

    char* w = (char*)d_ws;
    auto carve = [&](size_t bytes) -> char* {
        char* p = w; w += (bytes + 255) & ~(size_t)255; return p;
    };
    float*          h0     = (float*)carve(sizeof(float) * N_NODES * D);
    float*          agg    = (float*)carve(sizeof(float) * N_NODES * D);
    unsigned short* hbf    = (unsigned short*)carve(2ull * N_NODES * D);
    unsigned short* Wswz   = (unsigned short*)carve(2ull * N_RELS * D * D);
    int*            sorted = (int*)carve(sizeof(int) * N_EDGES);
    int*            counts = (int*)carve(sizeof(int) * N_RELS);
    int*            reloff = (int*)carve(sizeof(int) * (N_RELS + 1));
    int*            gtoff  = (int*)carve(sizeof(int) * (N_RELS + 1));
    int*            cursor = (int*)carve(sizeof(int) * N_RELS);
    int*            total  = (int*)carve(sizeof(int));
    float*          gpool  = (float*)carve(sizeof(float) * N_GRAPHS * D);

    const int ND = N_NODES * D;
    const int TB = 256;

    // counting sort of edges by relation (reused by all 3 layers)
    k_zero_i32<<<1, 128, 0, stream>>>(counts, N_RELS);
    k_zero_i32<<<1, 128, 0, stream>>>(cursor, N_RELS);
    k_hist   <<<(N_EDGES + TB - 1) / TB, TB, 0, stream>>>(etypes, counts);
    k_scan   <<<1, 1, 0, stream>>>(counts, reloff, gtoff, total);
    k_scatter<<<(N_EDGES + TB - 1) / TB, TB, 0, stream>>>(etypes, reloff, cursor, sorted);

    // initial bf16 features
    k_cvt_bf16<<<(ND + TB - 1) / TB, TB, 0, stream>>>(node_feats, hbf, ND);

    const int GROUP_MAX   = N_EDGES / (16 * GROUP) + N_RELS;   // 12565 upper bound
    const int RGCN_BLOCKS = (GROUP_MAX + 3) / 4;               // 4 waves / block

    for (int l = 0; l < 3; ++l) {
        k_swizzle_W<<<(N_RELS * D * D + TB - 1) / TB, TB, 0, stream>>>(Wr[l], Wswz);
        k_zero_f32<<<(ND + TB - 1) / TB, TB, 0, stream>>>(agg, ND);
        k_rgcn_wmma<<<RGCN_BLOCKS, 128, 0, stream>>>(hbf, Wswz, sorted, src, dst,
                                                     reloff, gtoff, total, agg);
        k_bias_relu_cvt<<<(ND + TB - 1) / TB, TB, 0, stream>>>(agg, br[l], h0, hbf);
    }

    // per-graph sum pooling + MLP head
    k_zero_f32<<<(N_GRAPHS * D + TB - 1) / TB, TB, 0, stream>>>(gpool, N_GRAPHS * D);
    k_pool<<<(ND + TB - 1) / TB, TB, 0, stream>>>(h0, gid, gpool);
    k_mlp<<<N_GRAPHS, D, 0, stream>>>(gpool, fcW[0], fcb[0], fcW[1], fcb[1],
                                      fcW[2], fcb[2], pW, pb, (float*)d_out);
}
